// EdgeRegressionModel_23570780521013
// MI455X (gfx1250) — matrix operand order repaired and checked by reading, not compiled, not run on hardware
//
#include <hip/hip_runtime.h>
#include <hip/hip_bf16.h>

// ---------------------------------------------------------------------------
// MI455X (gfx1250, wave32) implementation of the EdgeRegression GNN.
// All GEMMs use v_wmma_f32_16x16x32_bf16 (bf16 in, f32 accumulate).
// The E x (32x32) per-edge weight tensor is never materialized: the fused
// conv kernel generates 16x16 weight tiles in registers, applies BN+lrelu,
// multiplies by gathered x[src] (staged in LDS) and scatters with
// global_atomic_add_f32 into aggr[dst].
// B operands and conv-phase A operands are pre-swizzled into fragment order (bf16) so each such
// fragment is a single 32-byte vector load.
// ---------------------------------------------------------------------------

typedef __attribute__((ext_vector_type(16))) __bf16 v16bf;
typedef __attribute__((ext_vector_type(8)))  float  v8f;
typedef __attribute__((ext_vector_type(4)))  float  v4f;

static constexpr int Nn  = 10000;
static constexpr int Ee  = 160000;
static constexpr int NG  = 16;

// ---- workspace layout (floats) --------------------------------------------
static constexpr size_t OFF_STATS_A = 0;        // 64
static constexpr size_t OFF_AB_A    = 64;       // 64
static constexpr size_t OFF_STATS2  = 128;      // 2048
static constexpr size_t OFF_AB2     = 2176;     // 2048
static constexpr size_t OFF_XCSTAT  = 4224;     // 16
static constexpr size_t OFF_XAB     = 4240;     // 16
static constexpr size_t OFF_ECSTAT  = 4256;     // 20(+pad)
static constexpr size_t OFF_EAB     = 4288;     // 40
static constexpr size_t OFF_POOLS   = 4328;     // 512
static constexpr size_t OFF_POOLCNT = 4840;     // 16
static constexpr size_t OFF_POOLED  = 4856;     // 512
static constexpr size_t OFF_STATS1  = 5368;     // 2
static constexpr size_t OFF_AB1     = 5376;     // 2
static constexpr size_t OFF_WFRAG   = 5632;     // 1024 (layer-W bf16 B-frags, up to 128 v16bf)
static constexpr size_t OFF_XN      = 8192;                          // N*32 (padded z-scored x)
static constexpr size_t OFF_EAN    = OFF_XN   + (size_t)Nn * 32;     // E*32 (padded z-scored ea)
static constexpr size_t OFF_W2S    = OFF_EAN  + (size_t)Ee * 32;     // 16384 (W2 bf16 B-frags)
static constexpr size_t OFF_XE     = OFF_W2S  + 16384;               // N*32
static constexpr size_t OFF_EE     = OFF_XE   + (size_t)Nn * 32;     // E*32
static constexpr size_t OFF_H1     = OFF_EE   + (size_t)Ee * 32;     // N*32
static constexpr size_t OFF_H2     = OFF_H1   + (size_t)Nn * 32;     // N*32
static constexpr size_t OFF_AGGR   = OFF_H2   + (size_t)Nn * 32;     // N*32
static constexpr size_t OFF_G      = OFF_AGGR + (size_t)Nn * 32;     // N*32
static constexpr size_t OFF_TMPN   = OFF_G    + (size_t)Nn * 32;     // N*32
static constexpr size_t OFF_PRE    = OFF_TMPN + (size_t)Nn * 32;     // E*32
static constexpr size_t OFF_HIDB   = OFF_PRE  + (size_t)Ee * 32;     // E*32 (f32) / bf16 A-frags
// total ~22.7M floats (~91 MB) of d_ws

// ---------------------------------------------------------------------------
__device__ __forceinline__ float lrelu(float v) { return v > 0.f ? v : 0.01f * v; }

#define WMMA_BF16(A, B, C) \
  __builtin_amdgcn_wmma_f32_16x16x32_bf16(false, (A), false, (B), (short)0, (C), false, false)

// A-fragment (16x32 bf16) from a 32-col f32 row-major matrix: 4x b128 loads.
// lane L: row m = L&15, half = L>>4; element p -> K = ((p<8)?p:p+8)+8*half,
// i.e. runs [8h,8h+8) and [16+8h,24+8h).
__device__ __forceinline__ v16bf load_a32(const float* __restrict__ A, size_t row0, int lane) {
  int half = lane >> 4, m = lane & 15;
  const float* r = A + (row0 + (size_t)m) * 32 + 8 * half;
  v4f q0 = *(const v4f*)(r);
  v4f q1 = *(const v4f*)(r + 4);
  v4f q2 = *(const v4f*)(r + 16);
  v4f q3 = *(const v4f*)(r + 20);
  v16bf a;
#pragma unroll
  for (int j = 0; j < 4; ++j) {
    a[j]      = (__bf16)q0[j];
    a[4 + j]  = (__bf16)q1[j];
    a[8 + j]  = (__bf16)q2[j];
    a[12 + j] = (__bf16)q3[j];
  }
  return a;
}

// ---------------------------------------------------------------------------
__global__ void k_zero(float* __restrict__ p, size_t n) {
  for (size_t i = (size_t)blockIdx.x * blockDim.x + threadIdx.x; i < n;
       i += (size_t)gridDim.x * blockDim.x)
    p[i] = 0.f;
}

// per-column sum/sumsq, register accumulation (D compile-time)
template <int D>
__global__ void k_colstats(const float* __restrict__ a, int rows, float* __restrict__ stats) {
  __shared__ float ls[2 * D];
  for (int i = threadIdx.x; i < 2 * D; i += blockDim.x) ls[i] = 0.f;
  __syncthreads();
  float s[D], q[D];
#pragma unroll
  for (int c = 0; c < D; ++c) { s[c] = 0.f; q[c] = 0.f; }
  for (size_t r = (size_t)blockIdx.x * blockDim.x + threadIdx.x; r < (size_t)rows;
       r += (size_t)gridDim.x * blockDim.x) {
#pragma unroll
    for (int c = 0; c < D; ++c) {
      float v = a[r * D + c];
      s[c] += v; q[c] += v * v;
    }
  }
#pragma unroll
  for (int c = 0; c < D; ++c) { atomicAdd(&ls[c], s[c]); atomicAdd(&ls[D + c], q[c]); }
  __syncthreads();
  for (int i = threadIdx.x; i < 2 * D; i += blockDim.x) unsafeAtomicAdd(&stats[i], ls[i]);
}

__global__ void k_zscore_fin(const float* __restrict__ stats, int D, float R,
                             unsigned mask, float* __restrict__ ab) {
  int j = threadIdx.x;
  if (j < D) {
    float m = stats[j] / R;
    float v = stats[D + j] / R - m * m;
    if ((mask >> j) & 1u) {
      float a = 1.f / (sqrtf(fmaxf(v, 0.f)) + 1e-8f);
      ab[j] = a;
      ab[D + j] = -m * a;
    } else {
      ab[j] = 1.f;
      ab[D + j] = 0.f;
    }
  }
}

// z-score apply into zero-padded 32-col layout: out[r*32+c] (c<D data, else 0)
__global__ void k_zscore_apply(const float* __restrict__ in, const float* __restrict__ ab,
                               float* __restrict__ out, size_t rows, int D) {
  for (size_t i = (size_t)blockIdx.x * blockDim.x + threadIdx.x; i < rows * 32;
       i += (size_t)gridDim.x * blockDim.x) {
    int c = (int)(i & 31);
    size_t r = i >> 5;
    float v = 0.f;
    if (c < D) v = in[r * D + c] * ab[c] + ab[D + c];
    out[i] = v;
  }
}

// out[i] = act(in[i]*alpha[c] + beta[c]), c = i % nch
__global__ void k_affine_apply(const float* __restrict__ in, const float* __restrict__ ab,
                               float* __restrict__ out, size_t total, int nch, int act) {
  for (size_t i = (size_t)blockIdx.x * blockDim.x + threadIdx.x; i < total;
       i += (size_t)gridDim.x * blockDim.x) {
    int c = (int)(i % (size_t)nch);
    float v = in[i] * ab[c] + ab[nch + c];
    out[i] = act ? lrelu(v) : v;
  }
}

__global__ void k_bn_finalize(const float* __restrict__ stats, const float* __restrict__ g,
                              const float* __restrict__ be, float* __restrict__ ab,
                              int nch, float R) {
  int j = blockIdx.x * blockDim.x + threadIdx.x;
  if (j < nch) {
    float m = stats[j] / R;
    float v = stats[nch + j] / R - m * m;
    float rs = rsqrtf(v + 1e-5f);
    float a = g[j] * rs;
    ab[j] = a;
    ab[nch + j] = be[j] - m * a;
  }
}

// ---------------------------------------------------------------------------
// Swizzle a layer weight W[KACT x 32] (f32) into bf16 B-fragments:
// frag[(chunk*2+nt)*32 + lane][p] = W[chunk*32 + ((p<8)?p:p+8)+8*(lane>>4)][nt*16 + (lane&15)]
// (zero-padded beyond KACT). CHUNKS=1 covers K<=32, CHUNKS=2 covers K<=64.
template <int KACT, int CHUNKS>
__global__ void k_swizzle_wfrag(const float* __restrict__ Wm, v16bf* __restrict__ out) {
  int idx = blockIdx.x * blockDim.x + threadIdx.x;
  if (idx < CHUNKS * 2 * 32) {
    int lane = idx & 31, nt = (idx >> 5) & 1, chunk = idx >> 6;
    int half = lane >> 4, n = lane & 15;
    v16bf b;
#pragma unroll
    for (int p = 0; p < 16; ++p) {
      int k = chunk * 32 + ((p < 8) ? p : p + 8) + 8 * half;
      int ks = (k < KACT) ? k : 0;
      float v = Wm[(size_t)ks * 32 + nt * 16 + n];
      b[p] = (__bf16)((k < KACT) ? v : 0.f);
    }
    out[idx] = b;
  }
}

// ---------------------------------------------------------------------------
// Generic WMMA GEMM: pre[R x 32] = A[R x 32(padded)] @ W + bias, plus
// per-channel sum/sumsq stats. One wave per 16-row tile. B pre-swizzled.
__global__ void k_gemm32_stats(const float* __restrict__ A, const v16bf* __restrict__ Wf,
                               const float* __restrict__ bias, float* __restrict__ pre,
                               float* __restrict__ stats, int R) {
  __shared__ float lsum[32], lsq[32];
  if (threadIdx.x < 32) { lsum[threadIdx.x] = 0.f; lsq[threadIdx.x] = 0.f; }
  __syncthreads();
  int lane = threadIdx.x & 31, wid = threadIdx.x >> 5;
  int half = lane >> 4, n = lane & 15;
  int tile = blockIdx.x * (blockDim.x >> 5) + wid;
  if (tile * 16 < R) {
    size_t row0 = (size_t)tile * 16;
    v16bf a  = load_a32(A, row0, lane);
    v16bf b0 = Wf[lane];
    v16bf b1 = Wf[32 + lane];
    v8f acc0 = {}, acc1 = {};
    acc0 = WMMA_BF16(a, b0, acc0);
    acc1 = WMMA_BF16(a, b1, acc1);
    float bv0 = bias[n], bv1 = bias[16 + n];
    float s0 = 0.f, q0 = 0.f, s1 = 0.f, q1 = 0.f;
#pragma unroll
    for (int r = 0; r < 8; ++r) {
      int rm = r + 8 * half;
      float v0 = acc0[r] + bv0;
      float v1 = acc1[r] + bv1;
      pre[(row0 + rm) * 32 + n] = v0;
      pre[(row0 + rm) * 32 + 16 + n] = v1;
      s0 += v0; q0 += v0 * v0; s1 += v1; q1 += v1 * v1;
    }
    atomicAdd(&lsum[n], s0);      atomicAdd(&lsq[n], q0);
    atomicAdd(&lsum[16 + n], s1); atomicAdd(&lsq[16 + n], q1);
  }
  __syncthreads();
  if (threadIdx.x < 32) {
    unsafeAtomicAdd(&stats[threadIdx.x], lsum[threadIdx.x]);
    unsafeAtomicAdd(&stats[32 + threadIdx.x], lsq[threadIdx.x]);
  }
}

// Regressor layer 1: A row e = concat(h2[src[e]], h2[dst[e]]) (K = 64, 2 chunks)
__global__ void k_reg1_stats(const float* __restrict__ h2, const int* __restrict__ src,
                             const int* __restrict__ dst, const v16bf* __restrict__ Wf,
                             const float* __restrict__ bias, float* __restrict__ pre,
                             float* __restrict__ stats, int Etot) {
  __shared__ float lsum[32], lsq[32];
  if (threadIdx.x < 32) { lsum[threadIdx.x] = 0.f; lsq[threadIdx.x] = 0.f; }
  __syncthreads();
  int lane = threadIdx.x & 31, wid = threadIdx.x >> 5;
  int half = lane >> 4, n = lane & 15;
  int tile = blockIdx.x * (blockDim.x >> 5) + wid;
  if (tile * 16 < Etot) {
    size_t e0 = (size_t)tile * 16;
    int e = (int)e0 + (lane & 15);
    size_t se = (size_t)src[e], de = (size_t)dst[e];
    v8f acc0 = {}, acc1 = {};
#pragma unroll
    for (int chunk = 0; chunk < 2; ++chunk) {
      const float* row = chunk == 0 ? (h2 + se * 32) : (h2 + de * 32);
      const float* r = row + 8 * half;
      v4f q0 = *(const v4f*)(r);
      v4f q1 = *(const v4f*)(r + 4);
      v4f q2 = *(const v4f*)(r + 16);
      v4f q3 = *(const v4f*)(r + 20);
      v16bf a;
#pragma unroll
      for (int j = 0; j < 4; ++j) {
        a[j] = (__bf16)q0[j]; a[4 + j] = (__bf16)q1[j];
        a[8 + j] = (__bf16)q2[j]; a[12 + j] = (__bf16)q3[j];
      }
      v16bf b0 = Wf[chunk * 64 + lane];
      v16bf b1 = Wf[chunk * 64 + 32 + lane];
      acc0 = WMMA_BF16(a, b0, acc0);
      acc1 = WMMA_BF16(a, b1, acc1);
    }
    float bv0 = bias[n], bv1 = bias[16 + n];
    float s0 = 0.f, q0 = 0.f, s1 = 0.f, q1 = 0.f;
#pragma unroll
    for (int r = 0; r < 8; ++r) {
      int rm = r + 8 * half;
      float v0 = acc0[r] + bv0;
      float v1 = acc1[r] + bv1;
      pre[(e0 + rm) * 32 + n] = v0;
      pre[(e0 + rm) * 32 + 16 + n] = v1;
      s0 += v0; q0 += v0 * v0; s1 += v1; q1 += v1 * v1;
    }
    atomicAdd(&lsum[n], s0);      atomicAdd(&lsq[n], q0);
    atomicAdd(&lsum[16 + n], s1); atomicAdd(&lsq[16 + n], q1);
  }
  __syncthreads();
  if (threadIdx.x < 32) {
    unsafeAtomicAdd(&stats[threadIdx.x], lsum[threadIdx.x]);
    unsafeAtomicAdd(&stats[32 + threadIdx.x], lsq[threadIdx.x]);
  }
}

// ---------------------------------------------------------------------------
// W2 (32x1024 f32) -> bf16 B-fragments: frag[t][lane][p] = W2[((p<8)?p:p+8)+8*(lane>>4)][t*16+(lane&15)]
__global__ void k_swizzle_w2(const float* __restrict__ W2, v16bf* __restrict__ out) {
  int idx = blockIdx.x * blockDim.x + threadIdx.x;
  if (idx < 64 * 32) {
    int t = idx >> 5, lane = idx & 31;
    int half = lane >> 4, n = lane & 15;
    v16bf b;
#pragma unroll
    for (int p = 0; p < 16; ++p)
      b[p] = (__bf16)W2[(size_t)(((p < 8) ? p : p + 8) + 8 * half) * 1024 + t * 16 + n];
    out[idx] = b;
  }
}

// BN+lrelu applied to conv layer-1 pre (E x 32), written as bf16 A-fragments.
// One thread per edge: reads its 32-f32 row vectorized, emits 2x v16bf stores.
__global__ void k_apply_hidA(const float* __restrict__ pre, const float* __restrict__ ab,
                             v16bf* __restrict__ hidA, int Etot) {
  size_t e = (size_t)blockIdx.x * blockDim.x + threadIdx.x;
  if (e >= (size_t)Etot) return;
  const float* row = pre + e * 32;
  float val[32];
#pragma unroll
  for (int j = 0; j < 8; ++j) {
    v4f q = *(const v4f*)(row + 4 * j);
#pragma unroll
    for (int c = 0; c < 4; ++c) {
      int k = 4 * j + c;
      val[k] = lrelu(q[c] * ab[k] + ab[32 + k]);
    }
  }
  size_t tile = e >> 4;
  int m = (int)(e & 15);
#pragma unroll
  for (int half = 0; half < 2; ++half) {
    v16bf f;
#pragma unroll
    for (int p = 0; p < 16; ++p) f[p] = (__bf16)val[((p < 8) ? p : p + 8) + 8 * half];
    hidA[tile * 32 + m + 16 * half] = f;
  }
}

// ---------------------------------------------------------------------------
// Conv layer-2 stats pass: per-channel sum/sumsq of hidden @ W2 + b2 (1024 ch).
// 4 edge-tiles (64 edges) per wave; A and B fragments are single vector loads.
// All 4 WMMAs issue before any result is consumed (hazard distance covered).
__global__ void k_conv_stats2(const v16bf* __restrict__ hidA, const v16bf* __restrict__ W2s,
                              const float* __restrict__ b2, float* __restrict__ stats2,
                              int Etot) {
  __shared__ float ls[2048];
  for (int i = threadIdx.x; i < 2048; i += blockDim.x) ls[i] = 0.f;
  __syncthreads();
  int lane = threadIdx.x & 31, wid = threadIdx.x >> 5;
  int n = lane & 15;
  int grp = blockIdx.x * (blockDim.x >> 5) + wid;
  size_t e0 = (size_t)grp * 64;
  if (e0 < (size_t)Etot) {
    size_t tile0 = e0 >> 4;
    v16bf a[4];
#pragma unroll
    for (int t4 = 0; t4 < 4; ++t4) a[t4] = hidA[(tile0 + t4) * 32 + lane];
    for (int t = 0; t < 64; ++t) {
      v16bf b = W2s[t * 32 + lane];
      float bj = b2[t * 16 + n];
      v8f d[4];
#pragma unroll
      for (int t4 = 0; t4 < 4; ++t4) {
        v8f z = {};
        d[t4] = WMMA_BF16(a[t4], b, z);
      }
      float s = 0.f, q = 0.f;
#pragma unroll
      for (int t4 = 0; t4 < 4; ++t4)
#pragma unroll
        for (int r = 0; r < 8; ++r) {
          float v = d[t4][r] + bj;
          s += v; q += v * v;
        }
      atomicAdd(&ls[t * 16 + n], s);
      atomicAdd(&ls[1024 + t * 16 + n], q);
    }
  }
  __syncthreads();
  for (int i = threadIdx.x; i < 2048; i += blockDim.x) unsafeAtomicAdd(&stats2[i], ls[i]);
}

// finalize BN2, fold +b2 into beta: w = lrelu(acc*alpha + betaF)
__global__ void k_fin2(const float* __restrict__ stats2, const float* __restrict__ g2,
                       const float* __restrict__ be2, const float* __restrict__ b2,
                       float* __restrict__ ab2, float R) {
  int j = blockIdx.x * blockDim.x + threadIdx.x;
  if (j < 1024) {
    float m = stats2[j] / R;
    float v = stats2[1024 + j] / R - m * m;
    float rs = rsqrtf(v + 1e-5f);
    float a = g2[j] * rs;
    float be = be2[j] - m * a;
    ab2[j] = a;
    ab2[1024 + j] = be + b2[j] * a;
  }
}

// Fused: weight-gen WMMA -> BN+lrelu -> einsum with x[src] (LDS) -> atomic scatter.
__global__ void __launch_bounds__(128)
k_conv_fused(const v16bf* __restrict__ hidA, const v16bf* __restrict__ W2s,
             const float* __restrict__ ab2, const float* __restrict__ xin,
             const int* __restrict__ src, const int* __restrict__ dst,
             float* __restrict__ aggr, int Etot) {
  __shared__ float xs[4][64 * 32];   // 32 KB: 4 waves x 64 edges x 32 ch
  int lane = threadIdx.x & 31, wid = threadIdx.x >> 5;
  int half = lane >> 4, n = lane & 15;
  int grp = blockIdx.x * (blockDim.x >> 5) + wid;
  size_t e0 = (size_t)grp * 64;
  if (e0 >= (size_t)Etot) return;
  for (int el = 0; el < 64; ++el) {
    int s = src[e0 + el];
    xs[wid][el * 32 + lane] = xin[(size_t)s * 32 + lane];
  }
  __syncthreads();
  size_t tile0 = e0 >> 4;
  v16bf a[4];
#pragma unroll
  for (int t4 = 0; t4 < 4; ++t4) a[t4] = hidA[(tile0 + t4) * 32 + lane];
  v8f macc[4][2];
#pragma unroll
  for (int t4 = 0; t4 < 4; ++t4) { macc[t4][0] = (v8f){}; macc[t4][1] = (v8f){}; }
  for (int tt = 0; tt < 32; ++tt) {
#pragma unroll
    for (int ot = 0; ot < 2; ++ot) {
      int t = tt * 2 + ot;                 // i = tt, o-range = ot*16
      v16bf b = W2s[t * 32 + lane];
      float al = ab2[t * 16 + n];
      float bt = ab2[1024 + t * 16 + n];
      v8f d[4];
#pragma unroll
      for (int t4 = 0; t4 < 4; ++t4) {
        v8f z = {};
        d[t4] = WMMA_BF16(a[t4], b, z);
      }
#pragma unroll
      for (int t4 = 0; t4 < 4; ++t4)
#pragma unroll
        for (int r = 0; r < 8; ++r) {
          int m = r + 8 * half;
          float w = lrelu(d[t4][r] * al + bt);
          macc[t4][ot][r] += xs[wid][(t4 * 16 + m) * 32 + tt] * w;
        }
    }
  }
#pragma unroll
  for (int t4 = 0; t4 < 4; ++t4) {
#pragma unroll
    for (int r = 0; r < 8; ++r) {
      int m = r + 8 * half;
      int e = (int)e0 + t4 * 16 + m;
      int dn = dst[e];
      unsafeAtomicAdd(&aggr[(size_t)dn * 32 + n],      macc[t4][0][r]);
      unsafeAtomicAdd(&aggr[(size_t)dn * 32 + 16 + n], macc[t4][1][r]);
    }
  }
}

// out[node] = l2norm(aggr[node] + xin[node] @ root + bias); one wave per node
__global__ void k_root_l2norm(const float* __restrict__ aggr, const float* __restrict__ xin,
                              const float* __restrict__ root, const float* __restrict__ bias,
                              float* __restrict__ out, int Ntot) {
  int lane = threadIdx.x & 31;
  int node = blockIdx.x * (blockDim.x >> 5) + (threadIdx.x >> 5);
  if (node >= Ntot) return;
  float v = aggr[(size_t)node * 32 + lane] + bias[lane];
#pragma unroll
  for (int k = 0; k < 32; ++k) v += xin[(size_t)node * 32 + k] * root[k * 32 + lane];
  float nrm = v * v;
#pragma unroll
  for (int off = 16; off; off >>= 1) nrm += __shfl_xor(nrm, off, 32);
  nrm = fmaxf(sqrtf(nrm), 1e-12f);
  out[(size_t)node * 32 + lane] = v / nrm;
}

// regressor layer 2: pre[e] = hid[e] . w2 + b2, scalar sum/sumsq stats
__global__ void k_dot_stats(const float* __restrict__ hid, const float* __restrict__ w2,
                            const float* __restrict__ b2, float* __restrict__ pre,
                            float* __restrict__ stats, int Etot) {
  __shared__ float rs[256], rq[256];
  int tid = threadIdx.x;
  size_t e = (size_t)blockIdx.x * blockDim.x + tid;
  float s = 0.f, q = 0.f;
  if (e < (size_t)Etot) {
    float acc = b2[0];
#pragma unroll
    for (int j = 0; j < 8; ++j) {
      v4f hv = *(const v4f*)(hid + e * 32 + 4 * j);
      v4f wv = *(const v4f*)(w2 + 4 * j);
#pragma unroll
      for (int c = 0; c < 4; ++c) acc += hv[c] * wv[c];
    }
    pre[e] = acc;
    s = acc; q = acc * acc;
  }
  rs[tid] = s; rq[tid] = q;
  __syncthreads();
  for (int off = 128; off; off >>= 1) {
    if (tid < off) { rs[tid] += rs[tid + off]; rq[tid] += rq[tid + off]; }
    __syncthreads();
  }
  if (tid == 0) { unsafeAtomicAdd(&stats[0], rs[0]); unsafeAtomicAdd(&stats[1], rq[0]); }
}

__global__ void k_pool(const float* __restrict__ g, const int* __restrict__ batch,
                       float* __restrict__ sums, float* __restrict__ cnt, int Ntot) {
  size_t i = (size_t)blockIdx.x * blockDim.x + threadIdx.x;
  if (i < (size_t)Ntot * 32) {
    int node = (int)(i >> 5), c = (int)(i & 31);
    int b = batch[node];
    unsafeAtomicAdd(&sums[(size_t)b * 32 + c], g[i]);
    if (c == 0) unsafeAtomicAdd(&cnt[b], 1.f);
  }
}

__global__ void k_poolfin(const float* __restrict__ sums, const float* __restrict__ cnt,
                          float* __restrict__ pooled) {
  int j = blockIdx.x * blockDim.x + threadIdx.x;
  if (j < NG * 32) pooled[j] = sums[j] / fmaxf(cnt[j >> 5], 1.f);
}

// ---------------------------------------------------------------------------
static inline unsigned gb(size_t n, unsigned t) { return (unsigned)((n + t - 1) / t); }
static inline unsigned gbcap(size_t n, unsigned t, unsigned cap) {
  unsigned b = gb(n, t);
  return b > cap ? cap : b;
}

extern "C" void kernel_launch(void* const* d_in, const int* in_sizes, int n_in,
                              void* d_out, int out_size, void* d_ws, size_t ws_size,
                              hipStream_t stream) {
  // setup_inputs() insertion order, recursively flattened:
  // 0:x 1:edge_index 2:edge_attr 3:batch
  // 4..11 node_encode{w1,b1,g1,be1,w2,b2,g2,be2}, 12..19 edge_encode,
  // 20..27 conv1_nn, 28..35 conv2_nn, 36 conv1_root, 37 conv1_bias,
  // 38 conv2_root, 39 conv2_bias, 40..47 regressor, 48..55 graph_encode,
  // 56..63 graph_decode
  const float* x     = (const float*)d_in[0];
  const int*   ei    = (const int*)d_in[1];
  const float* ea    = (const float*)d_in[2];
  const int*   batch = (const int*)d_in[3];
  auto P = [&](int i) { return (const float*)d_in[i]; };
  const int* src = ei;
  const int* dst = ei + Ee;
  float* W = (float*)d_ws;
  float* out = (float*)d_out;

  float* statsA = W + OFF_STATS_A;
  float* abA    = W + OFF_AB_A;
  float* stats2 = W + OFF_STATS2;
  float* ab2    = W + OFF_AB2;
  float* PRE    = W + OFF_PRE;
  float* HIDB   = W + OFF_HIDB;
  v16bf* WFRAG  = (v16bf*)(W + OFF_WFRAG);
  v16bf* W2S    = (v16bf*)(W + OFF_W2S);
  v16bf* HIDA   = (v16bf*)(W + OFF_HIDB);   // bf16 A-frag view (conv phase only)

  auto zero = [&](float* p, size_t n) {
    k_zero<<<gbcap(n, 256, 4096), 256, 0, stream>>>(p, n);
  };

  // generic MLP layer: out = lrelu(BN(A @ w + b)); A padded to 32 cols
  auto mlp_layer = [&](const float* A, int K, int pbase, int which, float* outp, size_t R) {
    const float* w  = P(pbase + (which ? 4 : 0));
    const float* b  = P(pbase + (which ? 5 : 1));
    const float* g  = P(pbase + (which ? 6 : 2));
    const float* be = P(pbase + (which ? 7 : 3));
    if (K == 8)
      k_swizzle_wfrag<8, 1><<<1, 64, 0, stream>>>(w, WFRAG);
    else if (K == 10)
      k_swizzle_wfrag<10, 1><<<1, 64, 0, stream>>>(w, WFRAG);
    else
      k_swizzle_wfrag<32, 1><<<1, 64, 0, stream>>>(w, WFRAG);
    zero(statsA, 64);
    k_gemm32_stats<<<gb(R / 16, 8), 256, 0, stream>>>(A, WFRAG, b, PRE, statsA, (int)R);
    k_bn_finalize<<<1, 64, 0, stream>>>(statsA, g, be, abA, 32, (float)R);
    if (outp)
      k_affine_apply<<<gbcap(R * 32, 256, 8192), 256, 0, stream>>>(PRE, abA, outp, R * 32, 32, 1);
  };

  // ---- z-score normalize (into zero-padded 32-col layouts) -----------------
  zero(W + OFF_XCSTAT, 16);
  zero(W + OFF_ECSTAT, 20);
  k_colstats<8><<<gbcap(Nn, 256, 1024), 256, 0, stream>>>(x, Nn, W + OFF_XCSTAT);
  k_colstats<10><<<gbcap(Ee, 256, 2048), 256, 0, stream>>>(ea, Ee, W + OFF_ECSTAT);
  k_zscore_fin<<<1, 16, 0, stream>>>(W + OFF_XCSTAT, 8, (float)Nn, 0xC1u, W + OFF_XAB);
  k_zscore_fin<<<1, 16, 0, stream>>>(W + OFF_ECSTAT, 10, (float)Ee, 0x385u, W + OFF_EAB);
  k_zscore_apply<<<gbcap((size_t)Nn * 32, 256, 4096), 256, 0, stream>>>(x, W + OFF_XAB, W + OFF_XN, Nn, 8);
  k_zscore_apply<<<gbcap((size_t)Ee * 32, 256, 8192), 256, 0, stream>>>(ea, W + OFF_EAB, W + OFF_EAN, Ee, 10);

  // ---- encoders ------------------------------------------------------------
  mlp_layer(W + OFF_XN, 8, 4, 0, W + OFF_TMPN, Nn);       // node_encode L1
  mlp_layer(W + OFF_TMPN, 32, 4, 1, W + OFF_XE, Nn);      // node_encode L2 -> xe
  mlp_layer(W + OFF_EAN, 10, 12, 0, W + OFF_HIDB, Ee);    // edge_encode L1 (HIDB tmp)
  mlp_layer(W + OFF_HIDB, 32, 12, 1, W + OFF_EE, Ee);     // edge_encode L2 -> ee

  // ---- two NNConv layers ---------------------------------------------------
  for (int conv = 0; conv < 2; ++conv) {
    int pbase = conv ? 28 : 20;
    const float* xin  = conv ? (W + OFF_H1) : (W + OFF_XE);
    const float* root = P(conv ? 38 : 36);
    const float* cb   = P(conv ? 39 : 37);
    float* hout = conv ? (W + OFF_H2) : (W + OFF_H1);
    // swizzle W2 into bf16 B-fragments (64 KB, L2-resident)
    k_swizzle_w2<<<8, 256, 0, stream>>>(P(pbase + 4), W2S);
    // layer 1 of the edge-MLP: hidden = lrelu(BN(ee @ w1 + b1)), as bf16 A-frags
    mlp_layer(W + OFF_EE, 32, pbase, 0, nullptr, Ee);     // pre + BN coeffs only
    k_apply_hidA<<<gb(Ee, 256), 256, 0, stream>>>(PRE, abA, HIDA, Ee);
    // layer 2 stats pass (1024-channel BN over E)
    zero(stats2, 2048);
    k_conv_stats2<<<gb(Ee / 64, 4), 128, 0, stream>>>(HIDA, W2S, P(pbase + 5), stats2, Ee);
    k_fin2<<<4, 256, 0, stream>>>(stats2, P(pbase + 6), P(pbase + 7), P(pbase + 5), ab2, (float)Ee);
    // fused weight-gen -> einsum -> scatter
    zero(W + OFF_AGGR, (size_t)Nn * 32);
    k_conv_fused<<<gb(Ee / 64, 4), 128, 0, stream>>>(HIDA, W2S, ab2, xin, src, dst,
                                                     W + OFF_AGGR, Ee);
    k_root_l2norm<<<gb(Nn, 8), 256, 0, stream>>>(W + OFF_AGGR, xin, root, cb, hout, Nn);
  }

  // ---- edge regressor ------------------------------------------------------
  k_swizzle_wfrag<64, 2><<<1, 128, 0, stream>>>(P(40), WFRAG);
  zero(statsA, 64);
  k_reg1_stats<<<gb(Ee / 16, 8), 256, 0, stream>>>(W + OFF_H2, src, dst, WFRAG, P(41), PRE, statsA, Ee);
  k_bn_finalize<<<1, 64, 0, stream>>>(statsA, P(42), P(43), abA, 32, (float)Ee);
  k_affine_apply<<<gbcap((size_t)Ee * 32, 256, 8192), 256, 0, stream>>>(PRE, abA, HIDB, (size_t)Ee * 32, 32, 1);
  zero(W + OFF_STATS1, 2);
  k_dot_stats<<<Ee / 256, 256, 0, stream>>>(HIDB, P(44), P(45), PRE, W + OFF_STATS1, Ee);
  k_bn_finalize<<<1, 64, 0, stream>>>(W + OFF_STATS1, P(46), P(47), W + OFF_AB1, 1, (float)Ee);
  k_affine_apply<<<gbcap((size_t)Ee, 256, 4096), 256, 0, stream>>>(PRE, W + OFF_AB1, out, (size_t)Ee, 1, 1);

  // ---- graph head ----------------------------------------------------------
  mlp_layer(W + OFF_H2, 32, 48, 0, W + OFF_TMPN, Nn);     // graph_encode L1
  mlp_layer(W + OFF_TMPN, 32, 48, 1, W + OFF_G, Nn);      // graph_encode L2 -> g
  zero(W + OFF_POOLS, 512);
  zero(W + OFF_POOLCNT, 16);
  k_pool<<<gb((size_t)Nn * 32, 256), 256, 0, stream>>>(W + OFF_G, batch, W + OFF_POOLS, W + OFF_POOLCNT, Nn);
  k_poolfin<<<2, 256, 0, stream>>>(W + OFF_POOLS, W + OFF_POOLCNT, W + OFF_POOLED);
  mlp_layer(W + OFF_POOLED, 32, 56, 0, W + OFF_TMPN, NG); // graph_decode L1 (R=16)
  mlp_layer(W + OFF_TMPN, 32, 56, 1, out + Ee, NG);       // graph_decode L2 -> d_out tail
}
